// DeepEEGGCNN_75359496176059
// MI455X (gfx1250) — matrix operand order repaired and checked by reading, not compiled, not run on hardware
//
#include <hip/hip_runtime.h>
#include <hip/hip_fp16.h>

typedef __attribute__((ext_vector_type(16))) _Float16 v16h;
typedef __attribute__((ext_vector_type(8)))  _Float16 v8h;
typedef __attribute__((ext_vector_type(8)))  float    v8f;

#define NGRAPH 16384
#define NPG    8
#define NNODE  (NGRAPH * NPG)   // 131072
#define EPG    56               // fully-connected 8-node graph, no self loops

// ---------------------------------------------------------------------------
// Per-graph normalized adjacency (layer-invariant):
//   A[d][s] = dis[d]*w(s->d)*dis[s],  A[i][i] = 1/deg[i],  deg = 1 + sum_in w
// ---------------------------------------------------------------------------
__global__ void build_adj_kernel(const int* __restrict__ ei, const float* __restrict__ ea,
                                 float* __restrict__ adj, int E) {
  int g = blockIdx.x * blockDim.x + threadIdx.x;
  if (g >= NGRAPH) return;
  const int* srcv = ei;
  const int* dstv = ei + E;
  int base = g * EPG;
  float deg[NPG];
#pragma unroll
  for (int i = 0; i < NPG; i++) deg[i] = 1.0f;
  for (int k = 0; k < EPG; k++) deg[dstv[base + k] & 7] += ea[base + k];
  float dis[NPG];
#pragma unroll
  for (int i = 0; i < NPG; i++) dis[i] = rsqrtf(deg[i]);
  float* A = adj + g * 64;
#pragma unroll
  for (int i = 0; i < NPG; i++) A[i * 8 + i] = dis[i] * dis[i];
  for (int k = 0; k < EPG; k++) {
    int s = srcv[base + k] & 7;
    int d = dstv[base + k] & 7;
    A[d * 8 + s] = dis[s] * ea[base + k] * dis[d];
  }
}

// Pack block-diagonal 16x16 adjacency (2 graphs per mtile) into WMMA A-operand
// f16 layout: adjp[mtile][lane] = v8h holding this lane's K=8*half..8*half+7 slice.
__global__ void pack_adj_kernel(const float* __restrict__ adj, _Float16* __restrict__ adjp) {
  int t = blockIdx.x * blockDim.x + threadIdx.x;    // mtile*32 + lane
  if (t >= (NNODE / 16) * 32) return;
  int lane = t & 31, mtile = t >> 5;
  int half = lane >> 4, r = lane & 15;
  int g0 = mtile * 2;
  _Float16* dst = adjp + (size_t)t * 8;
#pragma unroll
  for (int j = 0; j < 4; j++)
#pragma unroll
    for (int e = 0; e < 2; e++) {
      int c = 2 * j + half * 8 + e;                 // K in 0..15
      float v = ((r >> 3) == (c >> 3))
                  ? adj[(g0 + (r >> 3)) * 64 + (r & 7) * 8 + (c & 7)] : 0.0f;
      dst[2 * j + e] = (_Float16)v;
    }
}

// Pack W (f32 [Din x Dout]) into B-operand f16 layout: Wp[(s*(Dout/16)+ntile)*32+lane] = v16h
__global__ void pack_w_kernel(const float* __restrict__ W, _Float16* __restrict__ Wp,
                              int Din, int Dout) {
  int t = blockIdx.x * blockDim.x + threadIdx.x;
  int total = ((Din + 31) / 32) * (Dout / 16) * 32;
  if (t >= total) return;
  int lane = t & 31, rest = t >> 5;
  int ntile = rest % (Dout / 16), s = rest / (Dout / 16);
  int half = lane >> 4, idx = lane & 15;
  int col = ntile * 16 + idx;
  _Float16* dst = Wp + (size_t)t * 16;
  for (int j = 0; j < 8; j++)
    for (int e = 0; e < 2; e++) {
      int kr = s * 32 + 8 * half + 2 * j + e + (j >= 4 ? 8 : 0);
      dst[2 * j + e] = (kr < Din) ? (_Float16)W[kr * Dout + col] : (_Float16)0.0f;
    }
}

// ---------------------------------------------------------------------------
// One GCN layer, fused:
//   A-load applies previous layer's BN (scale/shift) + LeakyReLU on the fly
//   WMMA #1: T = H @ W ; WMMA #2: Out = blockdiag(A) @ T ; + bias + BN stats.
// 4 waves per block, each owning one 16x16 output tile.
// ---------------------------------------------------------------------------
template <int Din, int Dout, bool FIRST>
__global__ void __launch_bounds__(128)
gcn_layer_kernel(const _Float16* __restrict__ hin, const float* __restrict__ xin,
                 const float* __restrict__ scIn, const float* __restrict__ shIn,
                 const _Float16* __restrict__ Wp, const float* __restrict__ bias,
                 const _Float16* __restrict__ adjp, _Float16* __restrict__ tout,
                 float* __restrict__ bnsum, float* __restrict__ bnsum2) {
  constexpr int KSTEPS = (Din + 31) / 32;
  __shared__ __align__(16) _Float16 Tl[4][256];     // column-major 16x16 tile per wave
  const int lane  = threadIdx.x;
  const int wv    = threadIdx.y;
  const int half  = lane >> 4;
  const int idx   = lane & 15;
  const int mtile = blockIdx.x * 4 + wv;
  const int ntile = blockIdx.y;
  const int row   = mtile * 16 + idx;
  const int col   = ntile * 16 + idx;

  // ---- WMMA #1: feature transform (input BN+lrelu fused into A load) ----
  v8f acc = {};
#pragma unroll
  for (int s = 0; s < KSTEPS; s++) {
    v16h a;
    if constexpr (FIRST) {                          // raw fp32 input, Din=6
      const float* xr = xin + (size_t)row * Din;
#pragma unroll
      for (int e = 0; e < 16; e++) a[e] = (_Float16)0.0f;
#pragma unroll
      for (int e = 0; e < Din; e++) {
        float v = xr[e];                            // unconditional load
        a[e] = half ? (_Float16)0.0f : (_Float16)v; // half1 K>=8 -> OOB pad
      }
    } else if constexpr (Din == 16) {
      const _Float16* hrow = hin + (size_t)row * Din;
      int kb = half * 8;
      v8h raw = *(const v8h*)(hrow + kb);
#pragma unroll
      for (int e = 0; e < 8; e++) {
        float y = scIn[kb + e] * (float)raw[e] + shIn[kb + e];
        y = (y >= 0.f) ? y : 0.01f * y;
        a[e]     = (_Float16)y;
        a[8 + e] = (_Float16)0.0f;                  // K 16..31 pad
      }
    } else {                                        // Din % 32 == 0
      const _Float16* hrow = hin + (size_t)row * Din;
      int kb0 = s * 32 + half * 8;
      int kb1 = s * 32 + 16 + half * 8;
      v8h r0 = *(const v8h*)(hrow + kb0);
      v8h r1 = *(const v8h*)(hrow + kb1);
#pragma unroll
      for (int e = 0; e < 8; e++) {
        float y0 = scIn[kb0 + e] * (float)r0[e] + shIn[kb0 + e];
        y0 = (y0 >= 0.f) ? y0 : 0.01f * y0;
        a[e] = (_Float16)y0;
        float y1 = scIn[kb1 + e] * (float)r1[e] + shIn[kb1 + e];
        y1 = (y1 >= 0.f) ? y1 : 0.01f * y1;
        a[8 + e] = (_Float16)y1;
      }
    }
    v16h b = ((const v16h*)Wp)[(s * (Dout / 16) + ntile) * 32 + lane];
    acc = __builtin_amdgcn_wmma_f32_16x16x32_f16(false, a, false, b, (short)0, acc, false, false);
  }

  // ---- relayout C -> B operand via LDS, column-major so DS ops are b128 ----
  // C: VGPR r => (row = r + 8*half, col = idx); store at col*16 + row.
  v8h tpk;
#pragma unroll
  for (int r = 0; r < 8; r++) tpk[r] = (_Float16)acc[r];
  *(v8h*)&Tl[wv][idx * 16 + half * 8] = tpk;
  __syncthreads();
  v8h t0 = *(const v8h*)&Tl[wv][idx * 16];          // rows 0..7  of col idx
  v8h t1 = *(const v8h*)&Tl[wv][idx * 16 + 8];      // rows 8..15 of col idx
  v16h tb;
#pragma unroll
  for (int e = 0; e < 8; e++) {
    tb[e]     = half ? t1[e] : t0[e];               // lane half h: K 8h..8h+7
    tb[8 + e] = (_Float16)0.0f;                     // K>=16 pad
  }

  // ---- WMMA #2: block-diagonal adjacency ----
  v16h aa;
  v8h al = *((const v8h*)adjp + (size_t)mtile * 32 + lane);
#pragma unroll
  for (int e = 0; e < 8; e++) { aa[e] = al[e]; aa[8 + e] = (_Float16)0.0f; }
  v8f oc = {};
  oc = __builtin_amdgcn_wmma_f32_16x16x32_f16(false, aa, false, tb, (short)0, oc, false, false);

  // ---- bias, store fp16 (pre-BN), BN statistics ----
  const float bch = bias[col];
  float s = 0.f, s2 = 0.f;
#pragma unroll
  for (int r = 0; r < 8; r++) {
    float v = oc[r] + bch;
    int node = mtile * 16 + r + 8 * half;
    tout[(size_t)node * Dout + col] = (_Float16)v;
    s += v; s2 += v * v;
  }
  s  += __shfl_xor(s, 16, 32);
  s2 += __shfl_xor(s2, 16, 32);
  if (half == 0) {
    atomicAdd(&bnsum[col], s);
    atomicAdd(&bnsum2[col], s2);
  }
}

__global__ void bn_finalize_kernel(const float* __restrict__ bnsum, const float* __restrict__ bnsum2,
                                   const float* __restrict__ g, const float* __restrict__ be,
                                   float* __restrict__ scale, float* __restrict__ shift, int Dout) {
  int c = threadIdx.x;
  if (c >= Dout) return;
  const float invN = 1.0f / (float)NNODE;
  float m  = bnsum[c] * invN;
  float v  = bnsum2[c] * invN - m * m;
  float iv = rsqrtf(v + 1e-5f);
  float sc = g[c] * iv;
  scale[c] = sc;
  shift[c] = be[c] - sc * m;
}

// Final layer: BN + lrelu + per-graph mean pool (8 nodes), fused.
__global__ void pool_bn_kernel(const _Float16* __restrict__ t, const float* __restrict__ scale,
                               const float* __restrict__ shift, float* __restrict__ pooled) {
  int i = blockIdx.x * blockDim.x + threadIdx.x;   // g*128 + c
  int g = i >> 7, c = i & 127;
  float sc = scale[c], sh = shift[c], s = 0.f;
#pragma unroll
  for (int r = 0; r < 8; r++) {
    float y = sc * (float)t[(size_t)(g * 8 + r) * 128 + c] + sh;
    s += (y >= 0.f) ? y : 0.01f * y;
  }
  pooled[i] = s * 0.125f;
}

__global__ void __launch_bounds__(32)
mlp_kernel(const float* __restrict__ pooled,
           const float* __restrict__ fc1w, const float* __restrict__ fc1b,
           const float* __restrict__ fc2w, const float* __restrict__ fc2b,
           const float* __restrict__ ow,   const float* __restrict__ ob,
           float* __restrict__ out) {
  int g = blockIdx.x;
  int t = threadIdx.x;
  __shared__ float z1[30], z2[20];
  const float* p = pooled + (size_t)g * 128;
  if (t < 30) {
    float a = fc1b[t];
    for (int k = 0; k < 128; k++) a += p[k] * fc1w[k * 30 + t];
    z1[t] = (a >= 0.f) ? a : 0.01f * a;
  }
  __syncthreads();
  if (t < 20) {
    float a = fc2b[t];
    for (int k = 0; k < 30; k++) a += z1[k] * fc2w[k * 20 + t];
    z2[t] = (a >= 0.f) ? a : 0.01f * a;
  }
  __syncthreads();
  if (t == 0) {
    float a = ob[0];
    for (int k = 0; k < 20; k++) a += z2[k] * ow[k];
    out[g] = a;
  }
}

// ---------------------------------------------------------------------------
template <int Din, int Dout, bool FIRST>
static void run_layer(const _Float16* hin, const float* xin,
                      const float* scIn, const float* shIn,
                      const float* W, const float* B_, const float* G_, const float* BE_,
                      const _Float16* adjp, _Float16* tcur, float* bnstats,
                      float* scOut, float* shOut, _Float16* Wp, hipStream_t stream) {
  int total = ((Din + 31) / 32) * (Dout / 16) * 32;
  pack_w_kernel<<<(total + 63) / 64, 64, 0, stream>>>(W, Wp, Din, Dout);
  hipMemsetAsync(bnstats, 0, 256 * sizeof(float), stream);
  gcn_layer_kernel<Din, Dout, FIRST><<<dim3(NNODE / 64, Dout / 16), dim3(32, 4), 0, stream>>>(
      hin, xin, scIn, shIn, Wp, B_, adjp, tcur, bnstats, bnstats + 128);
  bn_finalize_kernel<<<1, 128, 0, stream>>>(bnstats, bnstats + 128, G_, BE_, scOut, shOut, Dout);
}

extern "C" void kernel_launch(void* const* d_in, const int* in_sizes, int n_in,
                              void* d_out, int out_size, void* d_ws, size_t ws_size,
                              hipStream_t stream) {
  const float* x  = (const float*)d_in[0];
  const int*   ei = (const int*)d_in[1];
  const float* ea = (const float*)d_in[2];
  const float *W[5], *B[5], *G[5], *BE[5];
  for (int l = 0; l < 5; l++) {
    W[l]  = (const float*)d_in[4 + 4 * l + 0];
    B[l]  = (const float*)d_in[4 + 4 * l + 1];
    G[l]  = (const float*)d_in[4 + 4 * l + 2];
    BE[l] = (const float*)d_in[4 + 4 * l + 3];
  }
  const float* fc1w = (const float*)d_in[24];
  const float* fc1b = (const float*)d_in[25];
  const float* fc2w = (const float*)d_in[26];
  const float* fc2b = (const float*)d_in[27];
  const float* outw = (const float*)d_in[28];
  const float* outb = (const float*)d_in[29];
  float* out = (float*)d_out;
  const int E = in_sizes[2];

  // workspace layout
  char* ws = (char*)d_ws;
  const size_t ADJ_B  = (size_t)NGRAPH * 64 * sizeof(float);              // 4 MB
  const size_t ADJP_B = (size_t)(NNODE / 16) * 32 * 8 * sizeof(_Float16); // 4 MB
  const size_t T_B    = (size_t)NNODE * 128 * sizeof(_Float16);           // 32 MB each
  float*    adj     = (float*)ws;
  _Float16* adjp    = (_Float16*)(ws + ADJ_B);
  _Float16* tA      = (_Float16*)(ws + ADJ_B + ADJP_B);
  _Float16* tB      = (_Float16*)(ws + ADJ_B + ADJP_B + T_B);
  char*     tail    = ws + ADJ_B + ADJP_B + 2 * T_B;
  float*    bnstats = (float*)tail;                    // 256 floats
  float*    scsh    = bnstats + 256;                   // 5 layers x 512 floats
  _Float16* Wp      = (_Float16*)(tail + 16384);       // <= 16 KB
  float*    pooled  = (float*)(tail + 65536);          // NG*128 floats

  float *sc[5], *sh[5];
  for (int l = 0; l < 5; l++) { sc[l] = scsh + l * 512; sh[l] = sc[l] + 256; }

  build_adj_kernel<<<(NGRAPH + 63) / 64, 64, 0, stream>>>(ei, ea, adj, E);
  pack_adj_kernel<<<((NNODE / 16) * 32 + 255) / 256, 256, 0, stream>>>(adj, adjp);

  run_layer<6, 16, true>  (nullptr, x,    nullptr, nullptr, W[0], B[0], G[0], BE[0],
                           adjp, tA, bnstats, sc[0], sh[0], Wp, stream);
  run_layer<16, 16, false>(tA, nullptr, sc[0], sh[0], W[1], B[1], G[1], BE[1],
                           adjp, tB, bnstats, sc[1], sh[1], Wp, stream);
  run_layer<16, 32, false>(tB, nullptr, sc[1], sh[1], W[2], B[2], G[2], BE[2],
                           adjp, tA, bnstats, sc[2], sh[2], Wp, stream);
  run_layer<32, 64, false>(tA, nullptr, sc[2], sh[2], W[3], B[3], G[3], BE[3],
                           adjp, tB, bnstats, sc[3], sh[3], Wp, stream);
  run_layer<64, 128, false>(tB, nullptr, sc[3], sh[3], W[4], B[4], G[4], BE[4],
                            adjp, tA, bnstats, sc[4], sh[4], Wp, stream);

  pool_bn_kernel<<<(NGRAPH * 128) / 256, 256, 0, stream>>>(tA, sc[4], sh[4], pooled);
  mlp_kernel<<<NGRAPH, 32, 0, stream>>>(pooled, fc1w, fc1b, fc2w, fc2b, outw, outb, out);
}
